// Attention_32521492365784
// MI455X (gfx1250) — matrix operand order repaired and checked by reading, not compiled, not run on hardware
//
#include <hip/hip_runtime.h>
#include <hip/hip_bf16.h>

typedef __bf16 bf16_t;
typedef __attribute__((ext_vector_type(16))) __bf16 v16bf;
typedef __attribute__((ext_vector_type(8))) float v8f;
typedef int v4i_ __attribute__((vector_size(4 * sizeof(int))));

#define B_ 2
#define S_ 2048
#define D_ 2048
#define H_ 32
#define KV_ 8
#define HD_ 64

// ---------------------------------------------------------------------------
// Async global -> LDS copy (CDNA5 GLOBAL_LOAD_ASYNC_TO_LDS_B128, ASYNCcnt),
// with synchronous uint4 fallback if the builtin is unavailable.
// ---------------------------------------------------------------------------
#if __has_builtin(__builtin_amdgcn_global_load_async_to_lds_b128) && \
    __has_builtin(__builtin_amdgcn_s_wait_asynccnt)
#define USE_ASYNC_LDS 1
#else
#define USE_ASYNC_LDS 0
#endif

__device__ inline void cp16_g2l(bf16_t* dst_lds, const bf16_t* src_g) {
#if USE_ASYNC_LDS
  __builtin_amdgcn_global_load_async_to_lds_b128(
      (__attribute__((address_space(1))) v4i_*)(size_t)src_g,
      (__attribute__((address_space(3))) v4i_*)dst_lds, 0, 0);
#else
  *(uint4*)dst_lds = *(const uint4*)src_g;
#endif
}

__device__ inline void cp_g2l_wait() {
#if USE_ASYNC_LDS
  __builtin_amdgcn_s_wait_asynccnt(0);
#endif
}

// ---------------------------------------------------------------------------
// WMMA fragment loaders (CDNA5 wave32 layouts, 16-bit operands, 16x16x32).
// All loaders read two 8-wide k runs per lane (k = 8h.., 16+8h.., h = lane/16) -> lower to ds_load_b128 pairs.
// ---------------------------------------------------------------------------
// A (16x32, MxK) from row-major [m][k] tile.
__device__ inline v16bf frag_a(const bf16_t* base, int pitch, int lane) {
  const int r  = lane & 15;
  const int kh = (lane >> 4) << 3;  // 0 or 8
  const bf16_t* p = base + (size_t)r * pitch + kh;
  v16bf a;
#pragma unroll
  for (int h = 0; h < 8; ++h) a[h] = p[h];
#pragma unroll
  for (int h = 0; h < 8; ++h) a[8 + h] = p[16 + h];
  return a;
}

// B (32x16, KxN) built from X stored row-major [n][k] (i.e. B = X^T).
__device__ inline v16bf frag_bT(const bf16_t* base, int pitch, int lane) {
  const int n  = lane & 15;
  const int kb = (lane >> 4) << 3;
  const bf16_t* p = base + (size_t)n * pitch + kb;
  v16bf b;
#pragma unroll
  for (int h = 0; h < 8; ++h) b[h] = p[h];
#pragma unroll
  for (int h = 0; h < 8; ++h) b[8 + h] = p[16 + h];
  return b;
}

// ---------------------------------------------------------------------------
// GEMM: C[M,N] (f32) = A[M,K] x Bt^T, A row-major [M][K], Bt row-major [N][K]
// (weights pre-transposed). 128x128 block tile, 64-deep K panels (2 WMMA
// k-steps), 8 waves (4 M-groups x 2 N-groups), 2x4 accumulators per wave.
// ---------------------------------------------------------------------------
__global__ __launch_bounds__(256) void gemm_bf16_nt_kernel(
    const bf16_t* __restrict__ A, const bf16_t* __restrict__ Bt,
    float* __restrict__ C, int M, int N, int K) {
  __shared__ bf16_t As[128][64];
  __shared__ bf16_t Bts[128][64];
  const int t    = threadIdx.x;
  const int lane = t & 31;
  const int w    = t >> 5;
  const int wm   = w & 3;   // rows 32*wm .. +31
  const int wn   = w >> 2;  // cols 64*wn .. +63
  const int m0   = blockIdx.y * 128;
  const int n0   = blockIdx.x * 128;

  v8f acc[2][4] = {};

  const int cr = t >> 1;           // 0..127
  const int cc = (t & 1) << 5;     // 0 or 32 (32 contiguous bf16 = 4 x 16B)

  for (int k0 = 0; k0 < K; k0 += 64) {
    __syncthreads();
#pragma unroll
    for (int q = 0; q < 4; ++q) {
      cp16_g2l(&As[cr][cc + q * 8],  A  + (size_t)(m0 + cr) * K + k0 + cc + q * 8);
      cp16_g2l(&Bts[cr][cc + q * 8], Bt + (size_t)(n0 + cr) * K + k0 + cc + q * 8);
    }
    if (k0 + 64 < K) {  // pull next panels toward caches (global_prefetch_b8)
      __builtin_prefetch(A + (size_t)(m0 + cr) * K + (k0 + 64) + cc, 0, 1);
      __builtin_prefetch(Bt + (size_t)(n0 + cr) * K + (k0 + 64) + cc, 0, 1);
    }
    cp_g2l_wait();
    __syncthreads();

#pragma unroll
    for (int ks = 0; ks < 2; ++ks) {
      v16bf af[2];
#pragma unroll
      for (int i = 0; i < 2; ++i) af[i] = frag_a(&As[wm * 32 + i * 16][ks * 32], 64, lane);
      v16bf bf[4];
#pragma unroll
      for (int j = 0; j < 4; ++j) bf[j] = frag_bT(&Bts[wn * 64 + j * 16][ks * 32], 64, lane);
#pragma unroll
      for (int j = 0; j < 4; ++j)
#pragma unroll
        for (int i = 0; i < 2; ++i)
          acc[i][j] = __builtin_amdgcn_wmma_f32_16x16x32_bf16(
              false, af[i], false, bf[j], (short)0, acc[i][j], false, false);
    }
  }

  const int rl = (lane >> 4) << 3;
  const int cl = lane & 15;
#pragma unroll
  for (int i = 0; i < 2; ++i)
#pragma unroll
    for (int j = 0; j < 4; ++j)
#pragma unroll
      for (int v = 0; v < 8; ++v) {
        const int row = m0 + wm * 32 + i * 16 + rl + v;
        const int col = n0 + wn * 64 + j * 16 + cl;
        C[(size_t)row * N + col] = acc[i][j][v];
      }
}

// ---------------------------------------------------------------------------
// Elementwise helpers
// ---------------------------------------------------------------------------
__global__ void f32_to_bf16_kernel(const float* __restrict__ in,
                                   bf16_t* __restrict__ out, int n) {
  int i = blockIdx.x * blockDim.x + threadIdx.x;
  if (i < n) out[i] = (bf16_t)in[i];
}

// Weight convert + transpose: in fp32 [K][N] row-major -> out bf16 [N][K]
__global__ void w_transpose_bf16_kernel(const float* __restrict__ in,
                                        bf16_t* __restrict__ out, int K, int N) {
  int idx = blockIdx.x * blockDim.x + threadIdx.x;  // over N*K, k fastest
  if (idx >= N * K) return;
  const int k = idx % K;
  const int n = idx / K;
  out[(size_t)n * K + k] = (bf16_t)in[(size_t)k * N + n];
}

// RoPE on q half of QG buffer, relayout -> q_b[B,H,S,HD] bf16
__global__ void rope_q_kernel(const float* __restrict__ qg,
                              const float* __restrict__ cosb,
                              const float* __restrict__ sinb,
                              bf16_t* __restrict__ qb, int n) {
  int idx = blockIdx.x * blockDim.x + threadIdx.x;
  if (idx >= n) return;
  const int d  = idx & 63;
  const int hh = (idx >> 6) & 31;
  const int s  = (idx >> 11) & 2047;
  const int b  = idx >> 22;
  const size_t rowbase = ((size_t)b * S_ + s) * (size_t)(2 * H_ * HD_) + (size_t)hh * HD_;
  const float x = qg[rowbase + d];
  const float r = (d < 32) ? -qg[rowbase + d + 32] : qg[rowbase + d - 32];
  const float o = x * cosb[s * HD_ + d] + r * sinb[s * HD_ + d];
  qb[(((size_t)(b * H_ + hh)) * S_ + s) * HD_ + d] = (bf16_t)o;
}

// RoPE on k, relayout -> k_b[B,KV,S,HD] bf16
__global__ void rope_k_kernel(const float* __restrict__ kraw,
                              const float* __restrict__ cosb,
                              const float* __restrict__ sinb,
                              bf16_t* __restrict__ kb, int n) {
  int idx = blockIdx.x * blockDim.x + threadIdx.x;
  if (idx >= n) return;
  const int d  = idx & 63;
  const int kv = (idx >> 6) & 7;
  const int s  = (idx >> 9) & 2047;
  const int b  = idx >> 20;
  const size_t rowbase = ((size_t)b * S_ + s) * (size_t)(KV_ * HD_) + (size_t)kv * HD_;
  const float x = kraw[rowbase + d];
  const float r = (d < 32) ? -kraw[rowbase + d + 32] : kraw[rowbase + d - 32];
  const float o = x * cosb[s * HD_ + d] + r * sinb[s * HD_ + d];
  kb[(((size_t)(b * KV_ + kv)) * S_ + s) * HD_ + d] = (bf16_t)o;
}

// v fp32 [B*S, KV*HD] -> bf16 [B,KV,HD,S] (d-major so attention V tiles and
// fragments are contiguous along keys)
__global__ void v_pack_kernel(const float* __restrict__ vraw,
                              bf16_t* __restrict__ vbt, int n) {
  int idx = blockIdx.x * blockDim.x + threadIdx.x;
  if (idx >= n) return;
  const int d  = idx & 63;
  const int kv = (idx >> 6) & 7;
  const int s  = (idx >> 9) & 2047;
  const int b  = idx >> 20;
  const float x = vraw[((size_t)b * S_ + s) * (size_t)(KV_ * HD_) + (size_t)kv * HD_ + d];
  vbt[(((size_t)(b * KV_ + kv)) * HD_ + d) * S_ + s] = (bf16_t)x;
}

// ---------------------------------------------------------------------------
// Flash attention: one block = one (b,h) and 64 query rows. Online softmax
// over 32 key blocks of 64. QK^T and P.V through WMMA; sigmoid-gated output.
// ---------------------------------------------------------------------------
__global__ __launch_bounds__(256) void attn_kernel(
    const bf16_t* __restrict__ qb, const bf16_t* __restrict__ kbuf,
    const bf16_t* __restrict__ vbt, const float* __restrict__ qg,
    bf16_t* __restrict__ attnb) {
  const int q0  = blockIdx.x * 64;
  const int bh  = blockIdx.y;
  const int h   = bh & (H_ - 1);
  const int b   = bh >> 5;
  const int kvh = h >> 2;  // H/KV = 4
  const int t    = threadIdx.x;
  const int lane = t & 31;
  const int w    = t >> 5;
  const int wm   = w & 3;   // query rows 16*wm..+15
  const int wn   = w >> 2;  // cols 32*wn..+31 (keys for S, head-dim for O)

  __shared__ bf16_t Qs[64][64];   // [q][d]
  __shared__ bf16_t Ks[64][64];   // [key][d]
  __shared__ bf16_t Vt[64][64];   // [d][key]  (transposed for PV fragments)
  __shared__ bf16_t Ps[64][64];   // [q][key]
  __shared__ float  Ss[64][64];
  __shared__ float  rowf[64];

  // Load Q tile (64x64 bf16) via async LDS copy
  {
    const int row = t >> 2, col = (t & 3) << 4;
    const bf16_t* src = qb + (((size_t)(b * H_ + h)) * S_ + q0 + row) * HD_ + col;
    cp16_g2l(&Qs[row][col], src);
    cp16_g2l(&Qs[row][col + 8], src + 8);
  }

  v8f oacc[2] = {};
  float m_i = -1e30f, l_i = 0.0f;

  for (int kb = 0; kb < S_ / 64; ++kb) {
    __syncthreads();  // previous iteration done with Ks/Vt/Ps
    {
      const int row = t >> 2, col = (t & 3) << 4;
      // K tile: [key][d], contiguous in d
      const bf16_t* ksrc = kbuf +
          (((size_t)(b * KV_ + kvh)) * S_ + kb * 64 + row) * HD_ + col;
      cp16_g2l(&Ks[row][col], ksrc);
      cp16_g2l(&Ks[row][col + 8], ksrc + 8);
      // V tile: [d][key], source is d-major so contiguous in key
      const bf16_t* vsrc = vbt +
          (((size_t)(b * KV_ + kvh)) * HD_ + row) * S_ + kb * 64 + col;
      cp16_g2l(&Vt[row][col], vsrc);
      cp16_g2l(&Vt[row][col + 8], vsrc + 8);
    }
    cp_g2l_wait();
    __syncthreads();

    // S = Q K^T (64x64), contraction over HD=64 (2 WMMA steps)
    v8f sacc[2] = {};
#pragma unroll
    for (int ks = 0; ks < 2; ++ks) {
      v16bf aq = frag_a(&Qs[wm * 16][ks * 32], 64, lane);
      v16bf bk[2];
#pragma unroll
      for (int tc = 0; tc < 2; ++tc)
        bk[tc] = frag_bT(&Ks[wn * 32 + tc * 16][ks * 32], 64, lane);
#pragma unroll
      for (int tc = 0; tc < 2; ++tc)
        sacc[tc] = __builtin_amdgcn_wmma_f32_16x16x32_bf16(
            false, aq, false, bk[tc], (short)0, sacc[tc], false, false);
    }
    const int rl = (lane >> 4) << 3;
    const int cl = lane & 15;
#pragma unroll
    for (int tc = 0; tc < 2; ++tc)
#pragma unroll
      for (int v = 0; v < 8; ++v)
        Ss[wm * 16 + rl + v][wn * 32 + tc * 16 + cl] = sacc[tc][v] * 0.125f;  // 1/sqrt(64)
    __syncthreads();

    // Online softmax bookkeeping: thread t < 64 owns query row t
    if (t < 64) {
      float rmax = -1e30f;
      for (int j = 0; j < 64; ++j) rmax = fmaxf(rmax, Ss[t][j]);
      const float m_new = fmaxf(m_i, rmax);
      const float alpha = __expf(m_i - m_new);
      float sum = 0.0f;
      for (int j = 0; j < 64; ++j) {
        const float p = __expf(Ss[t][j] - m_new);
        sum += p;
        Ps[t][j] = (bf16_t)p;
      }
      l_i = l_i * alpha + sum;
      m_i = m_new;
      rowf[t] = alpha;
    }
    __syncthreads();

    // Rescale running O, then O += P V (contraction over 64 keys, 2 steps)
#pragma unroll
    for (int v = 0; v < 8; ++v) {
      const float al = rowf[wm * 16 + rl + v];
      oacc[0][v] *= al;
      oacc[1][v] *= al;
    }
#pragma unroll
    for (int ks = 0; ks < 2; ++ks) {
      v16bf ap = frag_a(&Ps[wm * 16][ks * 32], 64, lane);
      v16bf bv[2];
#pragma unroll
      for (int tc = 0; tc < 2; ++tc)
        bv[tc] = frag_bT(&Vt[wn * 32 + tc * 16][ks * 32], 64, lane);
#pragma unroll
      for (int tc = 0; tc < 2; ++tc)
        oacc[tc] = __builtin_amdgcn_wmma_f32_16x16x32_bf16(
            false, ap, false, bv[tc], (short)0, oacc[tc], false, false);
    }
  }

  __syncthreads();
  if (t < 64) rowf[t] = 1.0f / l_i;
  __syncthreads();

  const int rl = (lane >> 4) << 3;
  const int cl = lane & 15;
#pragma unroll
  for (int tc = 0; tc < 2; ++tc)
#pragma unroll
    for (int v = 0; v < 8; ++v) {
      const int row  = wm * 16 + rl + v;
      const int dcol = wn * 32 + tc * 16 + cl;
      const size_t tok = (size_t)b * S_ + q0 + row;
      float val = oacc[tc][v] * rowf[row];
      const float g = qg[tok * (size_t)(2 * H_ * HD_) + (H_ * HD_) + h * HD_ + dcol];
      val *= 1.0f / (1.0f + __expf(-g));
      attnb[tok * (size_t)(H_ * HD_) + h * HD_ + dcol] = (bf16_t)val;
    }
}

// ---------------------------------------------------------------------------
// Host-side orchestration
// ---------------------------------------------------------------------------
extern "C" void kernel_launch(void* const* d_in, const int* in_sizes, int n_in,
                              void* d_out, int out_size, void* d_ws, size_t ws_size,
                              hipStream_t stream) {
  (void)in_sizes; (void)n_in; (void)out_size; (void)ws_size;
  const float* X    = (const float*)d_in[0];
  const float* cosb = (const float*)d_in[1];
  const float* sinb = (const float*)d_in[2];
  const float* Wq   = (const float*)d_in[3];
  const float* Wk   = (const float*)d_in[4];
  const float* Wv   = (const float*)d_in[5];
  const float* Wo   = (const float*)d_in[6];
  float* out = (float*)d_out;

  char* ws = (char*)d_ws;
  size_t off = 0;
  auto carve = [&](size_t bytes) -> char* {
    off = (off + 255) & ~(size_t)255;
    char* p = ws + off;
    off += bytes;
    return p;
  };
  const size_t BS = (size_t)B_ * S_;  // 4096 tokens
  bf16_t* Xb   = (bf16_t*)carve(BS * D_ * 2);
  bf16_t* WqT  = (bf16_t*)carve((size_t)(2 * H_ * HD_) * D_ * 2);  // [4096][2048]
  bf16_t* WkT  = (bf16_t*)carve((size_t)(KV_ * HD_) * D_ * 2);     // [512][2048]
  bf16_t* WvT  = (bf16_t*)carve((size_t)(KV_ * HD_) * D_ * 2);
  bf16_t* WoT  = (bf16_t*)carve((size_t)D_ * (H_ * HD_) * 2);      // [2048][2048]
  float*  qgF  = (float*) carve(BS * (2 * H_ * HD_) * 4);
  float*  krawF= (float*) carve(BS * (KV_ * HD_) * 4);
  float*  vrawF= (float*) carve(BS * (KV_ * HD_) * 4);
  bf16_t* qb   = (bf16_t*)carve(BS * (H_ * HD_) * 2);
  bf16_t* kbuf = (bf16_t*)carve(BS * (KV_ * HD_) * 2);
  bf16_t* vbt  = (bf16_t*)carve(BS * (KV_ * HD_) * 2);
  bf16_t* attnb= (bf16_t*)carve(BS * (H_ * HD_) * 2);

  // Activations: plain convert
  {
    const int n = (int)(BS * D_);
    f32_to_bf16_kernel<<<(n + 255) / 256, 256, 0, stream>>>(X, Xb, n);
  }
  // Weights: convert + transpose to [N][K]
  auto wt = [&](const float* s, bf16_t* d, int K, int N) {
    const int n = K * N;
    w_transpose_bf16_kernel<<<(n + 255) / 256, 256, 0, stream>>>(s, d, K, N);
  };
  wt(Wq, WqT, D_, 2 * H_ * HD_);
  wt(Wk, WkT, D_, KV_ * HD_);
  wt(Wv, WvT, D_, KV_ * HD_);
  wt(Wo, WoT, H_ * HD_, D_);

  // Projections: qg [4096,4096], k/v [4096,512]
  gemm_bf16_nt_kernel<<<dim3((2 * H_ * HD_) / 128, BS / 128), 256, 0, stream>>>(
      Xb, WqT, qgF, (int)BS, 2 * H_ * HD_, D_);
  gemm_bf16_nt_kernel<<<dim3((KV_ * HD_) / 128, BS / 128), 256, 0, stream>>>(
      Xb, WkT, krawF, (int)BS, KV_ * HD_, D_);
  gemm_bf16_nt_kernel<<<dim3((KV_ * HD_) / 128, BS / 128), 256, 0, stream>>>(
      Xb, WvT, vrawF, (int)BS, KV_ * HD_, D_);

  const int nq = B_ * S_ * H_ * HD_;   // 8388608
  const int nk = B_ * S_ * KV_ * HD_;  // 2097152
  rope_q_kernel<<<nq / 256, 256, 0, stream>>>(qgF, cosb, sinb, qb, nq);
  rope_k_kernel<<<nk / 256, 256, 0, stream>>>(krawF, cosb, sinb, kbuf, nk);
  v_pack_kernel<<<nk / 256, 256, 0, stream>>>(vrawF, vbt, nk);

  attn_kernel<<<dim3(S_ / 64, B_ * H_), 256, 0, stream>>>(qb, kbuf, vbt, qgF, attnb);

  // Output projection: [4096,2048] x [2048,2048]^T(pre-transposed) -> d_out
  gemm_bf16_nt_kernel<<<dim3(D_ / 128, BS / 128), 256, 0, stream>>>(
      attnb, WoT, out, (int)BS, D_, H_ * HD_);
}
